// AttentionLayer_6012954214897
// MI455X (gfx1250) — compile-verified, matrix operand order checked by reading, not run on hardware
//
#include <hip/hip_runtime.h>
#include <hip/hip_bf16.h>

// ---------------------------------------------------------------------------
// Fused windowed attention layer for MI455X (gfx1250), wave32 + WMMA bf16.
//
// B=32, Tq=1024, Tk=1024, C=E=256. Window mask keeps only columns
// [last, last+win) of scores; exp(-1e30 - max) underflows to 0, so attn is
// zero outside the window. Single kernel, 64 rows per workgroup:
//   phase 0: NT-zero-fill this WG's 64 attn rows; stage window K cols / V rows
//   phase 1: X = Q@W_in^T + b_in            (v_wmma_f32_16x16x32_bf16)
//   phase 2: 3 window scores + softmax -> attn window cols
//   phase 3: mid = sqrt(Tk) * w @ V[window] (rank-3, reuses X LDS)
//   phase 4: out = (mid@W_out^T + b_out + Q) * sqrt(0.5)   (WMMA, NT stores)
// ---------------------------------------------------------------------------

typedef __bf16 bf16_t;
typedef __attribute__((ext_vector_type(16))) __bf16 v16bf;
typedef __attribute__((ext_vector_type(8)))  float  v8f;
typedef __attribute__((ext_vector_type(4)))  float  v4f;  // native vec for NT stores

#define B_    32
#define TQ    1024
#define TK    1024
#define CC    256
#define EE    256
#define MAXW  8
#define ROWS  64              // rows per workgroup
#define NEG_INF_F (-1e30f)

// Load 8 consecutive floats, convert to bf16 into v16bf elements [base, base+8).
__device__ __forceinline__ void cvt8(const float* p, v16bf& v, int base) {
  v4f u0 = *(const v4f*)(p);
  v4f u1 = *(const v4f*)(p + 4);
  v[base + 0] = (bf16_t)u0.x; v[base + 1] = (bf16_t)u0.y;
  v[base + 2] = (bf16_t)u0.z; v[base + 3] = (bf16_t)u0.w;
  v[base + 4] = (bf16_t)u1.x; v[base + 5] = (bf16_t)u1.y;
  v[base + 6] = (bf16_t)u1.z; v[base + 7] = (bf16_t)u1.w;
}

__global__ __launch_bounds__(256) void attn_fused_kernel(
    const float* __restrict__ q, const float* __restrict__ keys,
    const float* __restrict__ values, const unsigned char* __restrict__ maskp,
    const float* __restrict__ W_in, const float* __restrict__ b_in,
    const float* __restrict__ W_out, const float* __restrict__ b_out,
    const int* __restrict__ p_last, const int* __restrict__ p_win,
    float* __restrict__ out, float* __restrict__ attn) {
  __shared__ float X_lds[ROWS][260];     // X tile, reused as "mid" tile
  __shared__ float klds[MAXW][256];      // staged key columns (window)
  __shared__ float vlds[MAXW][256];      // staged value rows (window)
  __shared__ float sc_lds[ROWS][4][MAXW];// partial scores
  __shared__ float w_lds[ROWS][MAXW];    // softmax weights

  const int tid    = threadIdx.x;
  const int lane   = tid & 31;
  const int wave   = tid >> 5;
  const int wave_m = wave >> 2;  // 0..1 -> which 32-row half
  const int wave_n = wave & 3;   // 0..3 -> which 64-col strip
  const int row0   = blockIdx.x * ROWS;  // global row (b*Tq + t)
  const int b      = row0 >> 10;         // Tq = 1024
  const int last   = p_last[0];
  int wcount = p_win[0];
  if (wcount > MAXW) wcount = MAXW;
  if (last + wcount > TK) wcount = TK - last;  // window guarantees >= 1

  // ---- phase 0a: non-temporal zero-fill of this WG's attn rows ----
  {
    v4f* arow4 = (v4f*)(attn + (size_t)row0 * TK);
    const v4f z = {0.f, 0.f, 0.f, 0.f};
    const int n4 = ROWS * (TK / 4);  // 16384 x 16B
    for (int i = tid; i < n4; i += 256)
      __builtin_nontemporal_store(z, arow4 + i);
  }
  // ---- phase 0b: stage window key columns / value rows into LDS ----
  for (int idx = tid; idx < wcount * 256; idx += 256) {
    int j = idx >> 8, e = idx & 255;
    klds[j][e] = keys[(size_t)b * EE * TK + (size_t)e * TK + (last + j)];
    vlds[j][e] = values[(size_t)b * TK * EE + (size_t)(last + j) * EE + e];
  }

  // ---- phase 1: X = Q @ W_in^T  (WMMA bf16, f32 accumulate) ----
  // A frag (16x32 bf16): lane L -> row (L&15), K = kbase..+7 and kbase+16..+23
  const int kbase = (lane >> 4) * 8;
  const float* qrow0 = q + (size_t)(row0 + wave_m * 32 + (lane & 15)) * CC;
  const float* qrow1 = qrow0 + 16 * CC;  // second 16-row tile

  v8f zero8 = {};
  v8f acc[2][4];
#pragma unroll
  for (int mt = 0; mt < 2; ++mt)
#pragma unroll
    for (int nt = 0; nt < 4; ++nt) acc[mt][nt] = zero8;

  for (int k0 = 0; k0 < CC; k0 += 32) {
    v16bf a0, a1;
    cvt8(qrow0 + k0 + kbase,      a0, 0);
    cvt8(qrow0 + k0 + kbase + 16, a0, 8);
    cvt8(qrow1 + k0 + kbase,      a1, 0);
    cvt8(qrow1 + k0 + kbase + 16, a1, 8);
#pragma unroll
    for (int nt = 0; nt < 4; ++nt) {
      const int erow = wave_n * 64 + nt * 16 + (lane & 15);
      const float* pb = W_in + (size_t)erow * CC + k0 + kbase;
      if (k0 + 32 < CC) __builtin_prefetch((const void*)(pb + 32), 0, 1);
      v16bf bf;
      cvt8(pb, bf, 0);
      cvt8(pb + 16, bf, 8);
      acc[0][nt] = __builtin_amdgcn_wmma_f32_16x16x32_bf16(
          false, a0, false, bf, (short)0, acc[0][nt], false, false);
      acc[1][nt] = __builtin_amdgcn_wmma_f32_16x16x32_bf16(
          false, a1, false, bf, (short)0, acc[1][nt], false, false);
    }
  }
  // D layout: VGPR v, lane L -> M = v + 8*(L>>4), N = L&15
#pragma unroll
  for (int nt = 0; nt < 4; ++nt) {
    const int e = wave_n * 64 + nt * 16 + (lane & 15);
    const float bi = b_in[e];
#pragma unroll
    for (int mt = 0; mt < 2; ++mt)
#pragma unroll
      for (int v = 0; v < 8; ++v) {
        const int m = wave_m * 32 + mt * 16 + v + 8 * (lane >> 4);
        X_lds[m][e] = acc[mt][nt][v] + bi;
      }
  }
  __syncthreads();

  // ---- phase 2: window scores + softmax ----
  {
    const int r = tid >> 2, p = tid & 3;  // 64 rows x 4 partials
    float part[MAXW];
#pragma unroll
    for (int j = 0; j < MAXW; ++j) part[j] = 0.f;
    for (int e = p * 64; e < p * 64 + 64; ++e) {
      const float xv = X_lds[r][e];
#pragma unroll
      for (int j = 0; j < MAXW; ++j)
        if (j < wcount) part[j] += xv * klds[j][e];
    }
#pragma unroll
    for (int j = 0; j < MAXW; ++j)
      if (j < wcount) sc_lds[r][p][j] = part[j];
  }
  __syncthreads();
  if (tid < ROWS) {
    const int r = tid;
    float s[MAXW];
    float mx = NEG_INF_F;
#pragma unroll
    for (int j = 0; j < MAXW; ++j) {
      if (j < wcount) {
        float v = 0.f;
#pragma unroll
        for (int p = 0; p < 4; ++p) v += sc_lds[r][p][j];
        if (maskp[(size_t)b * TK + last + j]) v = NEG_INF_F;
        s[j] = v;
        mx = fmaxf(mx, v);
      }
    }
    float denom = 0.f;
#pragma unroll
    for (int j = 0; j < MAXW; ++j)
      if (j < wcount) { s[j] = __expf(s[j] - mx); denom += s[j]; }
    const float inv = 1.0f / denom;
    const size_t arow = (size_t)(row0 + r) * TK + last;
#pragma unroll
    for (int j = 0; j < MAXW; ++j) {
      if (j < wcount) {
        const float w = s[j] * inv;
        w_lds[r][j] = w;
        attn[arow + j] = w;  // overwrites zeros; ordered by __syncthreads fence
      }
    }
  }
  __syncthreads();

  // ---- phase 3: mid = sqrt(Tk) * w @ V[window]  (reuse X_lds) ----
  {
    const int e = tid;  // 0..255
    float vj[MAXW];
#pragma unroll
    for (int j = 0; j < MAXW; ++j)
      if (j < wcount) vj[j] = vlds[j][e];
    const float sqrt_s = 32.0f;  // sqrt(Tk), Tk = 1024
    for (int r = 0; r < ROWS; ++r) {
      float a2 = 0.f;
#pragma unroll
      for (int j = 0; j < MAXW; ++j)
        if (j < wcount) a2 += w_lds[r][j] * vj[j];
      X_lds[r][e] = a2 * sqrt_s;
    }
  }
  __syncthreads();

  // ---- phase 4: out = (mid @ W_out^T + b_out + residual) * sqrt(0.5) ----
#pragma unroll
  for (int mt = 0; mt < 2; ++mt)
#pragma unroll
    for (int nt = 0; nt < 4; ++nt) acc[mt][nt] = zero8;

  const float* ma0 = &X_lds[wave_m * 32 + (lane & 15)][0];
  const float* ma1 = ma0 + 16 * 260;
  for (int k0 = 0; k0 < EE; k0 += 32) {
    v16bf a0, a1;
    cvt8(ma0 + k0 + kbase,      a0, 0);
    cvt8(ma0 + k0 + kbase + 16, a0, 8);
    cvt8(ma1 + k0 + kbase,      a1, 0);
    cvt8(ma1 + k0 + kbase + 16, a1, 8);
#pragma unroll
    for (int nt = 0; nt < 4; ++nt) {
      const int crow = wave_n * 64 + nt * 16 + (lane & 15);
      const float* pb = W_out + (size_t)crow * EE + k0 + kbase;
      if (k0 + 32 < EE) __builtin_prefetch((const void*)(pb + 32), 0, 1);
      v16bf bf;
      cvt8(pb, bf, 0);
      cvt8(pb + 16, bf, 8);
      acc[0][nt] = __builtin_amdgcn_wmma_f32_16x16x32_bf16(
          false, a0, false, bf, (short)0, acc[0][nt], false, false);
      acc[1][nt] = __builtin_amdgcn_wmma_f32_16x16x32_bf16(
          false, a1, false, bf, (short)0, acc[1][nt], false, false);
    }
  }
  const float rsq = 0.70710678118654752f;  // sqrt(0.5)
#pragma unroll
  for (int nt = 0; nt < 4; ++nt) {
    const int c = wave_n * 64 + nt * 16 + (lane & 15);
    const float bo = b_out[c];
#pragma unroll
    for (int mt = 0; mt < 2; ++mt)
#pragma unroll
      for (int v = 0; v < 8; ++v) {
        const int m = wave_m * 32 + mt * 16 + v + 8 * (lane >> 4);
        const size_t gr = (size_t)(row0 + m);
        const float val = (acc[mt][nt][v] + bo + q[gr * CC + c]) * rsq;
        __builtin_nontemporal_store(val, &out[gr * CC + c]);
      }
  }
}

extern "C" void kernel_launch(void* const* d_in, const int* in_sizes, int n_in,
                              void* d_out, int out_size, void* d_ws, size_t ws_size,
                              hipStream_t stream) {
  const float* q             = (const float*)d_in[0];
  const float* keys          = (const float*)d_in[1];
  const float* vals          = (const float*)d_in[2];
  const unsigned char* mask  = (const unsigned char*)d_in[3];
  const float* W_in          = (const float*)d_in[4];
  const float* b_in          = (const float*)d_in[5];
  const float* W_out         = (const float*)d_in[6];
  const float* b_out         = (const float*)d_in[7];
  const int* p_last          = (const int*)d_in[8];
  const int* p_win           = (const int*)d_in[9];

  float* out  = (float*)d_out;
  float* attn = out + (size_t)B_ * TQ * CC;  // tuple: (out, attn) concatenated

  attn_fused_kernel<<<(B_ * TQ) / ROWS, 256, 0, stream>>>(
      q, keys, vals, mask, W_in, b_in, W_out, b_out, p_last, p_win, out, attn);
}
